// AdvancedSkipResidualBlock_61125974556804
// MI455X (gfx1250) — matrix operand order repaired and checked by reading, not compiled, not run on hardware
//
#include <hip/hip_runtime.h>
#include <math.h>

// ---------------------------------------------------------------------------
// Problem constants (reference setup_inputs)
// ---------------------------------------------------------------------------
static constexpr int BATCH = 8;
static constexpr int C     = 512;     // channels (Cin = Cout = S)
static constexpr int T     = 4096;    // timesteps
static constexpr int DIL   = 2;       // dilation; causal taps: t and t-2

static constexpr int TM = 128;        // M tile (output channels) per block
static constexpr int TN = 128;        // N tile (timesteps) per block
static constexpr int TK = 32;         // K step = wmma 16x16x32 depth
static constexpr int LDA = 40;        // LDS row stride (halfs): 80B -> conflict-free b128
static constexpr int LDB = 40;        // LDS col stride (halfs): 64B row + 16B pad
static constexpr int NTHREADS = 256;  // 8 wave32 waves

typedef _Float16 fp16_t;
typedef __attribute__((ext_vector_type(16))) _Float16 v16h;
typedef __attribute__((ext_vector_type(8)))  _Float16 v8h;
typedef __attribute__((ext_vector_type(4)))  _Float16 v4h;
typedef __attribute__((ext_vector_type(8)))  float    v8f;
typedef __attribute__((ext_vector_type(4)))  float    v4f;
typedef __attribute__((ext_vector_type(4)))  unsigned int u32x4;
typedef __attribute__((ext_vector_type(8)))  int      i32x8;
typedef __attribute__((ext_vector_type(4)))  int      i32x4;

#define WMMA_F16(A, B, CC) \
  __builtin_amdgcn_wmma_f32_16x16x32_f16(false, (A), false, (B), (short)0, (CC), false, false)

// C/D 16x16 f32 accumulator: element r -> M = 8*(lane>=16)+r, N = lane&15
// => per lane, the 8 accumulator rows are 8 CONSECUTIVE output channels.

// ---------------------------------------------------------------------------
// Fragment loads: contiguous 32B per lane (2x ds_load_b128), bank-conflict-free
// A is stored [row][k] in contiguous k order, so lane<16 reads
// k0..15 and lane>=16 reads k16..31 as one contiguous run.
// B is [col][k] col-major: lane<16 reads k0..15, lane>=16 reads k16..31.
// ---------------------------------------------------------------------------
__device__ inline v16h frag_from(const fp16_t* p) {
  const v8h lo = *(const v8h*)p;
  const v8h hi = *(const v8h*)(p + 8);
  v16h r;
#pragma unroll
  for (int h = 0; h < 8; ++h) { r[h] = lo[h]; r[h + 8] = hi[h]; }
  return r;
}
__device__ inline v16h frag_a(const fp16_t* As, int lane, int rowBase) {
  return frag_from(As + (rowBase + (lane & 15)) * LDA + ((lane >> 4) << 4));
}
__device__ inline v16h frag_b(const fp16_t* Bs, int lane, int col0) {
  return frag_from(Bs + (col0 + (lane & 15)) * LDB + ((lane >> 4) << 4));
}

// A tile fill: TM x TK from row-major fp32 (row stride strideK), float4 loads,
// packed f16 convert, 8B LDS stores at the contiguous k position.
__device__ inline void fill_a_tile(fp16_t* As, const float* __restrict__ src,
                                   int m0, int strideK, int kk0, int tid) {
#pragma unroll
  for (int j = 0; j < (TM * TK / 4) / NTHREADS; ++j) {   // 1024 quads / 256 thr
    const int f   = tid + j * NTHREADS;
    const int row = f >> 3;                               // 8 quads per 32-k row
    const int kq  = (f & 7) << 2;
    const v4f v = *(const v4f*)(src + (size_t)(m0 + row) * strideK + kk0 + kq);
    const int blk  = kq >> 3;
    const int pblk = blk;  // no swap: contiguous k order
    v4h h; h[0] = (fp16_t)v[0]; h[1] = (fp16_t)v[1];
           h[2] = (fp16_t)v[2]; h[3] = (fp16_t)v[3];
    *(v4h*)(As + row * LDA + (pblk << 3) + (kq & 7)) = h;
  }
}

// ---------------------------------------------------------------------------
// Tensor Data Mover: DMA a TN x TK f16 tile (row stride C halfs in global)
// into LDS laid out [col][k] with 64B rows + 16B pad (= LDB 40 halfs).
// D# per CDNA5 ISA. One issue per wave (wv==0
// gates to a single wave per block); completion via s_wait_tensorcnt.
// ---------------------------------------------------------------------------
__device__ inline void tdm_load_b_tile(fp16_t* dstLds, const fp16_t* srcGlobal) {
  const unsigned long long ga = (unsigned long long)(uintptr_t)srcGlobal;
  const unsigned int lds = (unsigned int)(uintptr_t)dstLds;   // low 32b = LDS offset

  u32x4 g0;
  g0[0] = 1u;                                             // count=1, user descriptor
  g0[1] = lds;                                            // lds_addr (bytes)
  g0[2] = (unsigned int)(ga & 0xFFFFFFFFu);               // global_addr[31:0]
  g0[3] = (unsigned int)((ga >> 32) & 0x01FFFFFFu)        // global_addr[56:32]
        | (2u << 30);                                     // type = 2 ("image")

  i32x8 g1;
  g1[0] = (int)((1u << 16)        // data_size = 1 -> 2 bytes
              | (1u << 20)        // pad_enable
              | (3u << 22)        // pad_interval code 3 -> every 16 DWORDs (64B row)
              | (3u << 25));      // pad_amount  code 3 -> 4 DWORDs (16B)
  g1[1] = (int)((unsigned)C << 16);                 // tensor_dim0[15:0] = 512
  g1[2] = (int)(((unsigned)T & 0xFFFFu) << 16);     // tensor_dim1[15:0] = 4096
  g1[3] = (int)((unsigned)TK << 16);                // tile_dim0 = 32
  g1[4] = TN;                                       // tile_dim1 = 128 (tile_dim2 = 0)
  g1[5] = C;                                        // tensor_dim0_stride[31:0] = 512
  g1[6] = 0;                                        // stride hi / dim1_stride lo
  g1[7] = 0;

  const i32x4 z4 = {};
#if __clang_major__ >= 23
  const i32x8 z8 = {};
  __builtin_amdgcn_tensor_load_to_lds(g0, g1, z4, z4, z8, 0);
#else
  __builtin_amdgcn_tensor_load_to_lds(g0, g1, z4, z4, 0);
#endif
}

// ---------------------------------------------------------------------------
// Kernel 1: xc_t[b][t][c] = (wc @ cond)[c,t] + bc[c] + x[b,c,t]
// GEMM M=C, N=B*T, K=C.  Transposed fp32 output -> vector epilogue stores.
// ---------------------------------------------------------------------------
__global__ __launch_bounds__(NTHREADS)
void k_condproj(const float* __restrict__ cond, const float* __restrict__ wc,
                const float* __restrict__ bc,   const float* __restrict__ x,
                float* __restrict__ xct) {
  __shared__ fp16_t As[2][TM * LDA];
  __shared__ fp16_t Bs[2][TN * LDB];
  const int tid = threadIdx.x, lane = tid & 31, wv = tid >> 5;
  const int m0 = blockIdx.y * TM, n0 = blockIdx.x * TN;
  const int b = n0 / T, t0 = n0 % T;

  auto stage = [&](int bufi, int kk0) {
    fill_a_tile(As[bufi], wc, m0, C, kk0, tid);
#pragma unroll
    for (int j = 0; j < 4; ++j) {                 // 32k x 32 col-quads = 1024
      const int q    = tid + j * NTHREADS;
      const int col4 = (q & 31) << 2;
      const int k    = q >> 5;
      const v4f v = *(const v4f*)(cond + ((size_t)b * C + kk0 + k) * T + t0 + col4);
#pragma unroll
      for (int u = 0; u < 4; ++u) Bs[bufi][(col4 + u) * LDB + k] = (fp16_t)v[u];
    }
  };

  v8f acc[8] = {};
  stage(0, 0);
  __syncthreads();
  constexpr int STEPS = C / TK;
  for (int s = 0; s < STEPS; ++s) {
    const int cur = s & 1;
    if (s + 1 < STEPS) stage(cur ^ 1, (s + 1) * TK);
    const v16h a = frag_a(As[cur], lane, wv * 16);
#pragma unroll
    for (int nt = 0; nt < 8; ++nt) {
      const v16h bq = frag_b(Bs[cur], lane, nt * 16);
      acc[nt] = WMMA_F16(a, bq, acc[nt]);
    }
    __syncthreads();
  }

  const int mf = m0 + wv * 16 + ((lane >> 4) << 3);   // 8 consecutive channels
  const v4f bc0 = *(const v4f*)(bc + mf);
  const v4f bc1 = *(const v4f*)(bc + mf + 4);
#pragma unroll
  for (int nt = 0; nt < 8; ++nt) {
    const int t = t0 + nt * 16 + (lane & 15);
    v4f o0, o1;
#pragma unroll
    for (int r = 0; r < 4; ++r) {
      o0[r] = acc[nt][r]     + bc0[r] + x[((size_t)b * C + mf + r)     * T + t];
      o1[r] = acc[nt][r + 4] + bc1[r] + x[((size_t)b * C + mf + 4 + r) * T + t];
    }
    float* dst = xct + ((size_t)b * T + t) * C + mf;
    *(v4f*)dst = o0;
    *(v4f*)(dst + 4) = o1;
  }
}

// ---------------------------------------------------------------------------
// Kernel 2: gated_t[b][t][c] = tanh(wf*z + bf) * sigmoid(wg*z + bg)
// z = [xc(t), xc(t-2)] tap-interleaved: K=2C, k = 2*ci + tap.
// wf/wg flat layout m*2C + 2*ci + tap is already row-major in k. f16 output.
// ---------------------------------------------------------------------------
__global__ __launch_bounds__(NTHREADS)
void k_gated(const float* __restrict__ xct, const float* __restrict__ wf,
             const float* __restrict__ bfv, const float* __restrict__ wg,
             const float* __restrict__ bgv, fp16_t* __restrict__ goutt) {
  __shared__ fp16_t AsF[2][TM * LDA];
  __shared__ fp16_t AsG[2][TM * LDA];
  __shared__ fp16_t Bs [2][TN * LDB];
  const int tid = threadIdx.x, lane = tid & 31, wv = tid >> 5;
  const int m0 = blockIdx.y * TM, n0 = blockIdx.x * TN;
  const int b = n0 / T, t0 = n0 % T;

  auto stage = [&](int bufi, int kk0) {
    fill_a_tile(AsF[bufi], wf, m0, 2 * C, kk0, tid);
    fill_a_tile(AsG[bufi], wg, m0, 2 * C, kk0, tid);
#pragma unroll
    for (int j = 0; j < 2; ++j) {                 // 128 cols x 4 k8-groups
      const int g   = tid + j * NTHREADS;
      const int col = g >> 2;
      const int k8  = (g & 3) << 3;
      const int t   = t0 + col;
      const int ci0 = (kk0 + k8) >> 1;            // 4 consecutive in-channels
      const v4f f1 = *(const v4f*)(xct + ((size_t)b * T + t) * C + ci0);  // tap1: x[t]
      v4f f0 = {};                                                         // tap0: x[t-2]
      if (t >= DIL) f0 = *(const v4f*)(xct + ((size_t)b * T + t - DIL) * C + ci0);
      v8h h;
#pragma unroll
      for (int u = 0; u < 4; ++u) { h[2 * u] = (fp16_t)f0[u]; h[2 * u + 1] = (fp16_t)f1[u]; }
      *(v8h*)(Bs[bufi] + col * LDB + k8) = h;
    }
  };

  v8f aF[8] = {};
  v8f aG[8] = {};
  stage(0, 0);
  __syncthreads();
  constexpr int STEPS = (2 * C) / TK;             // 32
  for (int s = 0; s < STEPS; ++s) {
    const int cur = s & 1;
    if (s + 1 < STEPS) stage(cur ^ 1, (s + 1) * TK);
    const v16h fa = frag_a(AsF[cur], lane, wv * 16);
    const v16h ga = frag_a(AsG[cur], lane, wv * 16);
#pragma unroll
    for (int nt = 0; nt < 8; ++nt) {
      const v16h bq = frag_b(Bs[cur], lane, nt * 16);
      aF[nt] = WMMA_F16(fa, bq, aF[nt]);
      aG[nt] = WMMA_F16(ga, bq, aG[nt]);
    }
    __syncthreads();
  }

  const int mf = m0 + wv * 16 + ((lane >> 4) << 3);
  const v4f bf0 = *(const v4f*)(bfv + mf), bf1 = *(const v4f*)(bfv + mf + 4);
  const v4f bg0 = *(const v4f*)(bgv + mf), bg1 = *(const v4f*)(bgv + mf + 4);
#pragma unroll
  for (int nt = 0; nt < 8; ++nt) {
    const int t = t0 + nt * 16 + (lane & 15);
    v8h o;
#pragma unroll
    for (int r = 0; r < 8; ++r) {
      const float bfr = (r < 4) ? bf0[r] : bf1[r - 4];
      const float bgr = (r < 4) ? bg0[r] : bg1[r - 4];
      const float fo = tanhf(aF[nt][r] + bfr);
      const float go = 1.0f / (1.0f + expf(-(aG[nt][r] + bgr)));
      o[r] = (fp16_t)(fo * go);
    }
    *(v8h*)(goutt + ((size_t)b * T + t) * C + mf) = o;   // one 16B store
  }
}

// ---------------------------------------------------------------------------
// Kernel 3: residual = wr@gated + br + xc ; skip = ws@gated + bs
// Dual-A GEMM. B tile comes via the Tensor Data Mover (f16, contiguous rows,
// hardware-padded into the LDB=40 LDS layout), overlapped with WMMA compute.
// ---------------------------------------------------------------------------
__global__ __launch_bounds__(NTHREADS)
void k_outproj(const fp16_t* __restrict__ goutt, const float* __restrict__ wr,
               const float* __restrict__ br,     const float* __restrict__ wsw,
               const float* __restrict__ bsv,    const float* __restrict__ xct,
               float* __restrict__ resid,        float* __restrict__ skip) {
  __shared__ fp16_t AsR[2][TM * LDA];
  __shared__ fp16_t AsS[2][TM * LDA];
  __shared__ fp16_t Bs [2][TN * LDB];
  const int tid = threadIdx.x, lane = tid & 31, wv = tid >> 5;
  const int m0 = blockIdx.y * TM, n0 = blockIdx.x * TN;
  const int b = n0 / T, t0 = n0 % T;

  auto stage = [&](int bufi, int kk0) {
    fill_a_tile(AsR[bufi], wr,  m0, C, kk0, tid);
    fill_a_tile(AsS[bufi], wsw, m0, C, kk0, tid);
    if (wv == 0) {   // one TDM issue per block; DMA overlaps WMMA on other buffer
      tdm_load_b_tile(Bs[bufi], goutt + ((size_t)b * T + t0) * C + kk0);
    }
  };

  v8f aR[8] = {};
  v8f aS[8] = {};
  stage(0, 0);
  if (wv == 0) __builtin_amdgcn_s_wait_tensorcnt(0);
  __syncthreads();
  constexpr int STEPS = C / TK;
  for (int s = 0; s < STEPS; ++s) {
    const int cur = s & 1;
    if (s + 1 < STEPS) stage(cur ^ 1, (s + 1) * TK);
    const v16h ra = frag_a(AsR[cur], lane, wv * 16);
    const v16h sa = frag_a(AsS[cur], lane, wv * 16);
#pragma unroll
    for (int nt = 0; nt < 8; ++nt) {
      const v16h bq = frag_b(Bs[cur], lane, nt * 16);
      aR[nt] = WMMA_F16(ra, bq, aR[nt]);
      aS[nt] = WMMA_F16(sa, bq, aS[nt]);
    }
    if (wv == 0) __builtin_amdgcn_s_wait_tensorcnt(0);  // next-buffer DMA done
    __syncthreads();                                    // publish to all waves
  }

  const int mf = m0 + wv * 16 + ((lane >> 4) << 3);
  const v4f br0 = *(const v4f*)(br + mf),  br1 = *(const v4f*)(br + mf + 4);
  const v4f bs0 = *(const v4f*)(bsv + mf), bs1 = *(const v4f*)(bsv + mf + 4);
#pragma unroll
  for (int nt = 0; nt < 8; ++nt) {
    const int t = t0 + nt * 16 + (lane & 15);
    const float* xp = xct + ((size_t)b * T + t) * C + mf;
    const v4f x0 = *(const v4f*)xp;
    const v4f x1 = *(const v4f*)(xp + 4);
#pragma unroll
    for (int r = 0; r < 8; ++r) {
      const size_t oidx = ((size_t)b * C + mf + r) * T + t;
      const float xv  = (r < 4) ? x0[r]  : x1[r - 4];
      const float brv = (r < 4) ? br0[r] : br1[r - 4];
      const float bsr = (r < 4) ? bs0[r] : bs1[r - 4];
      resid[oidx] = aR[nt][r] + brv + xv;
      skip[oidx]  = aS[nt][r] + bsr;
    }
  }
}

// ---------------------------------------------------------------------------
// Launch
// ---------------------------------------------------------------------------
extern "C" void kernel_launch(void* const* d_in, const int* in_sizes, int n_in,
                              void* d_out, int out_size, void* d_ws, size_t ws_size,
                              hipStream_t stream) {
  const float* x    = (const float*)d_in[0];
  const float* cond = (const float*)d_in[1];
  const float* wf   = (const float*)d_in[2];
  const float* bfv  = (const float*)d_in[3];
  const float* wg   = (const float*)d_in[4];
  const float* bgv  = (const float*)d_in[5];
  const float* wr   = (const float*)d_in[6];
  const float* br   = (const float*)d_in[7];
  const float* wsw  = (const float*)d_in[8];
  const float* bsv  = (const float*)d_in[9];
  const float* wc   = (const float*)d_in[10];
  const float* bc   = (const float*)d_in[11];

  const size_t NELEM = (size_t)BATCH * C * T;
  float*  resid = (float*)d_out;
  float*  skip  = resid + NELEM;

  float*  xct   = (float*)d_ws;                // [B][T][C] fp32 (64 MB)
  fp16_t* goutt = (fp16_t*)(xct + NELEM);      // [B][T][C] f16  (32 MB)

  dim3 grid((BATCH * T) / TN, C / TM);         // (256, 4)
  dim3 block(NTHREADS);

  k_condproj<<<grid, block, 0, stream>>>(cond, wc, bc, x, xct);
  k_gated   <<<grid, block, 0, stream>>>(xct, wf, bfv, wg, bgv, goutt);
  k_outproj <<<grid, block, 0, stream>>>(goutt, wr, br, wsw, bsv, xct, resid, skip);
}
